// StackedAdapter_50689204027520
// MI455X (gfx1250) — matrix operand order repaired and checked by reading, not compiled, not run on hardware
//
#include <hip/hip_runtime.h>
#include <hip/hip_bf16.h>

typedef __bf16 v16bf  __attribute__((ext_vector_type(16)));
typedef __bf16 bf16x8 __attribute__((ext_vector_type(8)));
typedef float  v8f    __attribute__((ext_vector_type(8)));
typedef int    v4i    __attribute__((ext_vector_type(4)));

#define D_DIM 1024
#define F_DIM 2048
#define G_DOM 4
#define S_LEN 512
#define PG    8
#define MG    (PG * S_LEN)     /* 4096 rows per domain group */
#define MTOT  (G_DOM * MG)     /* 16384 total rows           */
#define LN_EPS 1e-6f

// GEMM tiling: 256x128 block tile, 8 waves as 4(M)x2(N) -> 64x64 per wave,
// 16 accumulators/wave, K staged 64 deep, double-buffered LDS (96 KB).
#define BM 256
#define BN 128
#define KB 64
#define A_ELEMS (BM * KB)      /* 16384 bf16 = 32 KB per stage */
#define B_ELEMS (BN * KB)      /*  8192 bf16 = 16 KB per stage */

// ---------------------------------------------------------------------------
// CDNA5 async global->LDS copy path (ASYNCcnt-tracked), with sync fallback.
// Probe-learned signature: (int4* gaddr, int4* ldsaddr, imm offset, imm cpol).
// ---------------------------------------------------------------------------
#if defined(__has_builtin)
#  if __has_builtin(__builtin_amdgcn_global_load_async_to_lds_b128) && \
      __has_builtin(__builtin_amdgcn_s_wait_asynccnt)
#    define USE_ASYNC 1
#  endif
#endif
#ifndef USE_ASYNC
#  define USE_ASYNC 0
#endif

#if USE_ASYNC
// One imm offset is added to BOTH global and LDS addresses (ISA 15.18.3).
#define ASYNC_CP16(gp, lp, off)                                         \
  __builtin_amdgcn_global_load_async_to_lds_b128(                       \
      (v4i*)(gp), (v4i*)(lp), (off), 0)
#endif

// ---------------------------------------------------------------------------
// Tiled transpose + f32->bf16 convert: out[g][c][r] = bf16(in[g][r][c])
// ---------------------------------------------------------------------------
__global__ __launch_bounds__(256) void transpose_cvt_kernel(
    const float* __restrict__ in, __bf16* __restrict__ out, int R, int C) {
  __shared__ float tile[32][33];
  const int g = blockIdx.z;
  const float* A = in  + (size_t)g * R * C;
  __bf16*      B = out + (size_t)g * R * C;
  const int c  = blockIdx.x * 32 + threadIdx.x;
  const int r0 = blockIdx.y * 32;
#pragma unroll
  for (int i = threadIdx.y; i < 32; i += 8)
    tile[i][threadIdx.x] = A[(size_t)(r0 + i) * C + c];
  __syncthreads();
  const int oc = r0 + threadIdx.x;
  const int o0 = blockIdx.x * 32;
#pragma unroll
  for (int i = threadIdx.y; i < 32; i += 8)
    B[(size_t)(o0 + i) * R + oc] = (__bf16)tile[threadIdx.x][i];
}

// ---------------------------------------------------------------------------
// Gather (x[idx]) + LayerNorm (ddof=1, a2*(x-mu)/(std+eps)+b2) -> bf16 Xn,
// rows group-major: dest row = g*4096 + j*512 + s.
// ---------------------------------------------------------------------------
__global__ __launch_bounds__(256) void ln_gather_kernel(
    const float* __restrict__ x, const int* __restrict__ idx,
    const float* __restrict__ Gw, const float* __restrict__ Bn,
    __bf16* __restrict__ Xn) {
  const int row    = blockIdx.x;
  const int g      = row >> 12;
  const int within = row & (MG - 1);
  const int jb     = within >> 9;
  const int s      = within & (S_LEN - 1);
  const int src_b  = idx[g * PG + jb];
  const float* xr  = x + ((size_t)src_b * S_LEN + s) * D_DIM;

  const int tid = threadIdx.x;
  float vals[4], s1 = 0.f, s2 = 0.f;
#pragma unroll
  for (int i = 0; i < 4; ++i) {
    float v = xr[tid + i * 256];
    vals[i] = v; s1 += v; s2 += v * v;
  }
  __shared__ float rs[256], rq[256];
  rs[tid] = s1; rq[tid] = s2;
  __syncthreads();
#pragma unroll
  for (int off = 128; off > 0; off >>= 1) {
    if (tid < off) { rs[tid] += rs[tid + off]; rq[tid] += rq[tid + off]; }
    __syncthreads();
  }
  __shared__ float mu_s, inv_s;
  if (tid == 0) {
    float mu  = rs[0] * (1.0f / D_DIM);
    float var = (rq[0] - (float)D_DIM * mu * mu) * (1.0f / (D_DIM - 1));
    var  = fmaxf(var, 0.0f);
    mu_s = mu;
    inv_s = 1.0f / (sqrtf(var) + LN_EPS);
  }
  __syncthreads();
  const float mu = mu_s, inv = inv_s;
  const float* gg = Gw + (size_t)g * D_DIM;
  const float* bb = Bn + (size_t)g * D_DIM;
  __bf16* o = Xn + (size_t)row * D_DIM;
#pragma unroll
  for (int i = 0; i < 4; ++i) {
    int d = tid + i * 256;
    o[d] = (__bf16)(gg[d] * (vals[i] - mu) * inv + bb[d]);
  }
}

// ---------------------------------------------------------------------------
// WMMA fragment loads from LDS tiles (CDNA5 16-bit layouts, wave32).
// A tile: [BM][KB] row-major (m,k). B tile: [BN][KB] N-major (n,k).
// ---------------------------------------------------------------------------
__device__ __forceinline__ v16bf load_frag_a(const __bf16* base, int row, int kk, int hl) {
  union { v16bf v; bf16x8 h[2]; } u;
  const __bf16* p = base + row * KB + kk + hl * 16; // half0: K 0..15
  u.h[0] = *(const bf16x8*)(p);                     // half1: K 16..31
  u.h[1] = *(const bf16x8*)(p + 8);
  return u.v;
}
__device__ __forceinline__ v16bf load_frag_b(const __bf16* base, int col, int kk, int hl) {
  union { v16bf v; bf16x8 h[2]; } u;
  const __bf16* p = base + col * KB + kk + hl * 16; // half0: K 0..15
  u.h[0] = *(const bf16x8*)(p);                     // half1: K 16..31
  u.h[1] = *(const bf16x8*)(p + 8);
  return u.v;
}

// ---------------------------------------------------------------------------
// Stage copy: A tile BM x KB (128 B/row, 1 row/thread), B tile BN x KB
// (64 B half-row/thread). Async path: 12 B128 async ops per thread/stage.
// ---------------------------------------------------------------------------
template<int K>
__device__ __forceinline__ void issue_stage(const __bf16* __restrict__ Ag,
                                            const __bf16* __restrict__ Bg,
                                            __bf16* AsB, __bf16* BsB,
                                            int kt, int tid) {
  const __bf16* ga = Ag + (size_t)tid * K + kt;
  __bf16*       la = AsB + tid * KB;
  const int br = tid >> 1, bo = (tid & 1) * 32;
  const __bf16* gb = Bg + (size_t)br * K + kt + bo;
  __bf16*       lb = BsB + br * KB + bo;
#if USE_ASYNC
  ASYNC_CP16(ga, la, 0);   ASYNC_CP16(ga, la, 16);
  ASYNC_CP16(ga, la, 32);  ASYNC_CP16(ga, la, 48);
  ASYNC_CP16(ga, la, 64);  ASYNC_CP16(ga, la, 80);
  ASYNC_CP16(ga, la, 96);  ASYNC_CP16(ga, la, 112);
  ASYNC_CP16(gb, lb, 0);   ASYNC_CP16(gb, lb, 16);
  ASYNC_CP16(gb, lb, 32);  ASYNC_CP16(gb, lb, 48);
#else
  *(v16bf*)(la)      = *(const v16bf*)(ga);
  *(v16bf*)(la + 16) = *(const v16bf*)(ga + 16);
  *(v16bf*)(la + 32) = *(const v16bf*)(ga + 32);
  *(v16bf*)(la + 48) = *(const v16bf*)(ga + 48);
  *(v16bf*)(lb)      = *(const v16bf*)(gb);
  *(v16bf*)(lb + 16) = *(const v16bf*)(gb + 16);
#endif
}

// ---------------------------------------------------------------------------
// Shared GEMM core: C(256x128) += A(256xK) * B^T(128 rows of K), bf16 WMMA,
// double-buffered LDS, async prefetch one stage ahead.
// ---------------------------------------------------------------------------
template<int K>
__device__ __forceinline__ void gemm_core(const __bf16* __restrict__ Ag,
                                          const __bf16* __restrict__ Bg,
                                          __bf16* As, __bf16* Bs,
                                          v8f (&acc)[4][4]) {
  const int tid  = threadIdx.x;
  const int lane = tid & 31, wave = tid >> 5;
  const int waveM = wave >> 1, waveN = wave & 1;
  const int l15 = lane & 15, hl = lane >> 4;

  issue_stage<K>(Ag, Bg, As, Bs, 0, tid);
  constexpr int NS = K / KB;
  for (int s = 0; s < NS; ++s) {
    const int cur = s & 1;
    __syncthreads();                       // buffer (cur^1) free to overwrite
    if (s + 1 < NS) {
      issue_stage<K>(Ag, Bg, As + (cur ^ 1) * A_ELEMS, Bs + (cur ^ 1) * B_ELEMS,
                     (s + 1) * KB, tid);
#if USE_ASYNC
      __builtin_amdgcn_s_wait_asynccnt(12);  // stage s landed (12 newer pending)
#endif
    }
#if USE_ASYNC
    else {
      __builtin_amdgcn_s_wait_asynccnt(0);
    }
#endif
    __syncthreads();                       // stage s visible to all waves
    const __bf16* Ab = As + cur * A_ELEMS;
    const __bf16* Bb = Bs + cur * B_ELEMS;
#pragma unroll
    for (int kk = 0; kk < KB; kk += 32) {
      v16bf af[4], bfrag[4];
#pragma unroll
      for (int mi = 0; mi < 4; ++mi)
        af[mi] = load_frag_a(Ab, waveM * 64 + mi * 16 + l15, kk, hl);
#pragma unroll
      for (int ni = 0; ni < 4; ++ni)
        bfrag[ni] = load_frag_b(Bb, waveN * 64 + ni * 16 + l15, kk, hl);
#pragma unroll
      for (int mi = 0; mi < 4; ++mi)
#pragma unroll
        for (int ni = 0; ni < 4; ++ni)
          acc[mi][ni] = __builtin_amdgcn_wmma_f32_16x16x32_bf16(
              false, af[mi], false, bfrag[ni], (short)0, acc[mi][ni], false, false);
    }
  }
}

// ---------------------------------------------------------------------------
// GEMM1: H[g] = relu(Xn[g] (4096x1024) @ W1[g] (1024x2048) + B1[g]) -> bf16.
// ---------------------------------------------------------------------------
__global__ __launch_bounds__(256) void gemm1_relu_kernel(
    const __bf16* __restrict__ Xn, const __bf16* __restrict__ W1t,
    const float* __restrict__ B1, __bf16* __restrict__ H) {
  constexpr int K = D_DIM, N = F_DIM;
  __shared__ __bf16 As[2 * A_ELEMS];
  __shared__ __bf16 Bs[2 * B_ELEMS];
  const int g       = blockIdx.z;
  const int rowBase = g * MG + blockIdx.y * BM;
  const int colBase = blockIdx.x * BN;
  const __bf16* Ag = Xn  + (size_t)rowBase * K;
  const __bf16* Bg = W1t + (size_t)g * N * K + (size_t)colBase * K;

  v8f acc[4][4];
  const v8f vzero = {0.f,0.f,0.f,0.f,0.f,0.f,0.f,0.f};
#pragma unroll
  for (int mi = 0; mi < 4; ++mi)
#pragma unroll
    for (int ni = 0; ni < 4; ++ni) acc[mi][ni] = vzero;

  gemm_core<K>(Ag, Bg, As, Bs, acc);

  const int lane = threadIdx.x & 31, wave = threadIdx.x >> 5;
  const int waveM = wave >> 1, waveN = wave & 1;
  const int l15 = lane & 15, hl = lane >> 4;
  const float* bias = B1 + (size_t)g * N;
#pragma unroll
  for (int mi = 0; mi < 4; ++mi) {
#pragma unroll
    for (int ni = 0; ni < 4; ++ni) {
      const int col = colBase + waveN * 64 + ni * 16 + l15;
      const float b = bias[col];
      const int m0 = rowBase + waveM * 64 + mi * 16 + 8 * hl;
#pragma unroll
      for (int j = 0; j < 8; ++j) {
        float v = fmaxf(acc[mi][ni][j] + b, 0.0f);
        H[(size_t)(m0 + j) * N + col] = (__bf16)v;
      }
    }
  }
}

// ---------------------------------------------------------------------------
// GEMM2: out[idx] = x[idx] + (H[g] (4096x2048) @ W2[g] (2048x1024) + B2[g]).
// ---------------------------------------------------------------------------
__global__ __launch_bounds__(256) void gemm2_residual_kernel(
    const __bf16* __restrict__ H, const __bf16* __restrict__ W2t,
    const float* __restrict__ B2, const float* __restrict__ x,
    const int* __restrict__ idx, float* __restrict__ out) {
  constexpr int K = F_DIM, N = D_DIM;
  __shared__ __bf16 As[2 * A_ELEMS];
  __shared__ __bf16 Bs[2 * B_ELEMS];
  const int g       = blockIdx.z;
  const int rowBase = g * MG + blockIdx.y * BM;
  const int colBase = blockIdx.x * BN;
  const __bf16* Ag = H   + (size_t)rowBase * K;
  const __bf16* Bg = W2t + (size_t)g * N * K + (size_t)colBase * K;

  v8f acc[4][4];
  const v8f vzero = {0.f,0.f,0.f,0.f,0.f,0.f,0.f,0.f};
#pragma unroll
  for (int mi = 0; mi < 4; ++mi)
#pragma unroll
    for (int ni = 0; ni < 4; ++ni) acc[mi][ni] = vzero;

  gemm_core<K>(Ag, Bg, As, Bs, acc);

  const int lane = threadIdx.x & 31, wave = threadIdx.x >> 5;
  const int waveM = wave >> 1, waveN = wave & 1;
  const int l15 = lane & 15, hl = lane >> 4;
  const float* bias = B2 + (size_t)g * N;
#pragma unroll
  for (int mi = 0; mi < 4; ++mi) {
    const int rg0 = blockIdx.y * BM + waveM * 64 + mi * 16 + 8 * hl;
#pragma unroll
    for (int j = 0; j < 8; ++j) {
      const int rg = rg0 + j;
      const int jb = rg >> 9;
      const int sq = rg & (S_LEN - 1);
      const int ob = idx[g * PG + jb];
      const size_t orow = (size_t)ob * S_LEN + sq;
#pragma unroll
      for (int ni = 0; ni < 4; ++ni) {
        const int col = colBase + waveN * 64 + ni * 16 + l15;
        float v = acc[mi][ni][j] + bias[col];
        out[orow * N + col] = x[orow * N + col] + v;
      }
    }
  }
}

// ---------------------------------------------------------------------------
// Host launcher. Workspace (128 MB):
//   [0,16M)    W1t bf16 [4][2048][1024]
//   [16M,32M)  W2t bf16 [4][1024][2048]
//   [32M,64M)  Xn  bf16 [16384][1024]
//   [64M,128M) H   bf16 [16384][2048]
// ---------------------------------------------------------------------------
extern "C" void kernel_launch(void* const* d_in, const int* in_sizes, int n_in,
                              void* d_out, int out_size, void* d_ws, size_t ws_size,
                              hipStream_t stream) {
  const float* x  = (const float*)d_in[0];
  const int*  idx = (const int*)d_in[1];
  const float* W1 = (const float*)d_in[2];
  const float* B1 = (const float*)d_in[3];
  const float* W2 = (const float*)d_in[4];
  const float* B2 = (const float*)d_in[5];
  const float* Gw = (const float*)d_in[6];
  const float* Bn = (const float*)d_in[7];
  float* out = (float*)d_out;

  char* ws = (char*)d_ws;
  const size_t wbytes = (size_t)G_DOM * F_DIM * D_DIM * sizeof(__bf16); // 16 MB
  __bf16* W1t = (__bf16*)(ws);
  __bf16* W2t = (__bf16*)(ws + wbytes);
  __bf16* Xn  = (__bf16*)(ws + 2 * wbytes);
  __bf16* Hb  = (__bf16*)(ws + 2 * wbytes + (size_t)MTOT * D_DIM * sizeof(__bf16));

  transpose_cvt_kernel<<<dim3(F_DIM / 32, D_DIM / 32, G_DOM), dim3(32, 8), 0, stream>>>(
      W1, W1t, D_DIM, F_DIM);
  transpose_cvt_kernel<<<dim3(D_DIM / 32, F_DIM / 32, G_DOM), dim3(32, 8), 0, stream>>>(
      W2, W2t, F_DIM, D_DIM);
  ln_gather_kernel<<<MTOT, 256, 0, stream>>>(x, idx, Gw, Bn, Xn);
  gemm1_relu_kernel<<<dim3(F_DIM / BN, MG / BM, G_DOM), 256, 0, stream>>>(
      Xn, W1t, B1, Hb);
  gemm2_residual_kernel<<<dim3(D_DIM / BN, MG / BM, G_DOM), 256, 0, stream>>>(
      Hb, W2t, B2, x, idx, out);
}
